// EnhancedGAT_77747497992438
// MI455X (gfx1250) — compile-verified, matrix operand order checked by reading, not run on hardware
//
#include <hip/hip_runtime.h>
#include <math.h>

#define HID   128
#define HEADS 4
#define CH    32
#define LAYERS 3
#define NCLS  8
#define TSH   64
#define POOLD 320
#define NEG_SLOPE 0.2f
#define RPB   64    // rows per block in the WMMA GEMM

typedef __attribute__((ext_vector_type(16))) _Float16 v16h;
typedef __attribute__((ext_vector_type(8)))  float    v8f;

union FragU { v16h v; float4 f[2]; };

// float atomic max via sign-split integer atomics (init must be -inf-ish)
__device__ __forceinline__ void atomicMaxF(float* addr, float val) {
    if (val >= 0.0f) atomicMax((int*)addr, __float_as_int(val));
    else             atomicMin((unsigned int*)addr, (unsigned int)__float_as_int(val));
}

// ---------------- node encoder: h = relu(x @ enc_w + enc_b) ----------------
__global__ void k_encoder(const float* __restrict__ x, const float* __restrict__ w,
                          const float* __restrict__ b, float* __restrict__ h, int n_nodes) {
    int idx = blockIdx.x * blockDim.x + threadIdx.x;
    if (idx >= n_nodes * HID) return;
    int n = idx / HID, c = idx % HID;
    float acc = b[c];
#pragma unroll
    for (int k = 0; k < 4; k++) acc += x[n * 4 + k] * w[k * HID + c];
    h[idx] = fmaxf(acc, 0.0f);
}

// ---------------- hh = h @ W  (N x 128 @ 128 x 128), f16 WMMA, f32 accum ----
// Block: 128 threads = 4 waves, RPB=64 rows. Wave wv owns rows [wv*16, wv*16+16)
// and sweeps all 8 column tiles. W is staged into LDS in *fragment-major*
// order so each B fragment is two contiguous 16-byte LDS loads.
__global__ void __launch_bounds__(128)
k_gemm_wmma(const float* __restrict__ A, const float* __restrict__ W,
            float* __restrict__ out, int n_nodes) {
    __shared__ __align__(16) _Float16 sA[RPB * HID];          // 16 KB
    __shared__ __align__(16) _Float16 sBf[8 * 4 * 32 * 16];   // 32 KB fragment-major
    const int tid  = threadIdx.x;
    const int row0 = blockIdx.x * RPB;
    const bool fullblock = (row0 + RPB <= n_nodes);

    // stage W: coalesced global reads, swizzled LDS writes.
    // dst index for W[k][n]:  t=n>>4, l16=n&15, kt=k>>5, r=k&31;
    //   r<16 : half=r>>3,      i=r&7
    //   r>=16: half=(r-16)>>3, i=8+((r-16)&7)
    //   lane = half*16+l16 ; dst = (((t*4+kt)*32+lane)<<4) + i
#pragma unroll 4
    for (int j = 0; j < (HID * HID) / 128; j++) {
        int f = j * 128 + tid;
        float w = W[f];
        int k = f >> 7, n = f & 127;
        int t = n >> 4, l16 = n & 15;
        int kt = k >> 5, r = k & 31;
        int half, i;
        if (r < 16) { half = r >> 3; i = r & 7; }
        else        { half = (r - 16) >> 3; i = 8 + ((r - 16) & 7); }
        int lane = half * 16 + l16;
        sBf[(((t * 4 + kt) * 32 + lane) << 4) + i] = (_Float16)w;
    }
    // stage activation stripe (f32 -> f16), coalesced; uniform fast path
    if (fullblock) {
#pragma unroll 4
        for (int j = 0; j < RPB; j++)
            sA[j * HID + tid] = (_Float16)A[(size_t)(row0 + j) * HID + tid];
    } else {
        for (int j = 0; j < RPB; j++) {
            int r = row0 + j;
            sA[j * HID + tid] = (r < n_nodes) ? (_Float16)A[(size_t)r * HID + tid]
                                              : (_Float16)0.0f;
        }
    }
    __syncthreads();

    const int lane = tid & 31;
    const int wv   = tid >> 5;
    const int half = lane >> 4;
    const int l16  = lane & 15;
    const int arow = wv * 16 + l16;        // A row within block stripe

    // A fragments for this wave's stripe: resident across all 8 tiles
    FragU afrag[4];
#pragma unroll
    for (int kt = 0; kt < 4; kt++) {
        int kb = kt * 32;
        afrag[kt].f[0] = *(const float4*)(sA + arow * HID + kb +      half * 8);
        afrag[kt].f[1] = *(const float4*)(sA + arow * HID + kb + 16 + half * 8);
    }

    const int outrow0 = row0 + wv * 16;
#pragma unroll
    for (int t = 0; t < 8; t++) {
        v8f acc = {};
#pragma unroll
        for (int kt = 0; kt < 4; kt++) {
            const _Float16* bp = sBf + (((t * 4 + kt) * 32 + lane) << 4);
            FragU ub;
            ub.f[0] = *(const float4*)(bp);
            ub.f[1] = *(const float4*)(bp + 8);
            acc = __builtin_amdgcn_wmma_f32_16x16x32_f16(
                false, afrag[kt].v, false, ub.v, (short)0, acc, false, false);
        }
        // C/D: lane -> col (l16), VGPR v -> row v + 8*half.
        // Base pointer once; rows at stride HID. Uniform guard only for the
        // (single) partial tail block.
        float* obase = out + (size_t)(outrow0 + 8 * half) * HID + t * 16 + l16;
        if (fullblock) {
#pragma unroll
            for (int v = 0; v < 8; v++) obase[(size_t)v * HID] = acc[v];
        } else {
#pragma unroll
            for (int v = 0; v < 8; v++) {
                int r = outrow0 + v + 8 * half;
                if (r < n_nodes) out[(size_t)r * HID + t * 16 + l16] = acc[v];
            }
        }
    }
}

// ---------------- per-node attention coefficients a_s, a_d -----------------
__global__ void k_attn_coef(const float* __restrict__ hh, const float* __restrict__ as_,
                            const float* __restrict__ ad_, float* __restrict__ a_s,
                            float* __restrict__ a_d, int n_nodes) {
    int idx = blockIdx.x * blockDim.x + threadIdx.x;
    if (idx >= n_nodes * HEADS) return;
    int n = idx / HEADS, hd = idx % HEADS;
    const float* base = hh + (size_t)n * HID + hd * CH;
    float ss = 0.f, sd = 0.f;
#pragma unroll
    for (int c = 0; c < CH; c++) {
        float v = base[c];
        ss += v * as_[hd * CH + c];
        sd += v * ad_[hd * CH + c];
    }
    a_s[idx] = ss;
    a_d[idx] = sd;
}

// ---------------- per-layer scratch init -----------------------------------
__global__ void k_init_layer(float* __restrict__ outacc, float* __restrict__ emax,
                             float* __restrict__ esum, int n_nodes) {
    int idx = blockIdx.x * blockDim.x + threadIdx.x;
    if (idx < n_nodes * HID) outacc[idx] = 0.0f;
    if (idx < n_nodes * HEADS) { emax[idx] = -3.0e38f; esum[idx] = 0.0f; }
}

// ---------------- edge pass 1: segment max of leaky_relu scores ------------
__global__ void k_edge_max(const int* __restrict__ ei, int E, int n_nodes,
                           const float* __restrict__ a_s, const float* __restrict__ a_d,
                           float* __restrict__ emax) {
    int e = blockIdx.x * blockDim.x + threadIdx.x;
    int Etot = E + n_nodes;
    if (e >= Etot) return;
    int s = (e < E) ? ei[e]     : (e - E);   // self-loops appended
    int d = (e < E) ? ei[E + e] : (e - E);
#pragma unroll
    for (int hd = 0; hd < HEADS; hd++) {
        float v = a_s[s * HEADS + hd] + a_d[d * HEADS + hd];
        v = (v > 0.f) ? v : NEG_SLOPE * v;
        atomicMaxF(&emax[d * HEADS + hd], v);
    }
}

// ---------------- edge pass 2: exp(e - max) and segment sum ----------------
__global__ void k_edge_expsum(const int* __restrict__ ei, int E, int n_nodes,
                              const float* __restrict__ a_s, const float* __restrict__ a_d,
                              const float* __restrict__ emax, float* __restrict__ ew,
                              float* __restrict__ esum) {
    int e = blockIdx.x * blockDim.x + threadIdx.x;
    int Etot = E + n_nodes;
    if (e >= Etot) return;
    int s = (e < E) ? ei[e]     : (e - E);
    int d = (e < E) ? ei[E + e] : (e - E);
#pragma unroll
    for (int hd = 0; hd < HEADS; hd++) {
        float v = a_s[s * HEADS + hd] + a_d[d * HEADS + hd];
        v = (v > 0.f) ? v : NEG_SLOPE * v;
        float w = __expf(v - emax[d * HEADS + hd]);
        ew[(size_t)e * HEADS + hd] = w;
        atomicAdd(&esum[d * HEADS + hd], w);
    }
}

// ---------------- edge pass 3: out[dst] += hh[src] * alpha  (1 wave/edge) --
__global__ void __launch_bounds__(256)
k_edge_aggregate(const int* __restrict__ ei, int E, int n_nodes,
                 const float* __restrict__ hh, const float* __restrict__ ew,
                 const float* __restrict__ esum, float* __restrict__ outacc) {
    int wave = (blockIdx.x * blockDim.x + threadIdx.x) >> 5;
    int lane = threadIdx.x & 31;
    int Etot = E + n_nodes;
    if (wave >= Etot) return;
    int e = wave;
    int s = (e < E) ? ei[e]     : (e - E);
    int d = (e < E) ? ei[E + e] : (e - E);
    int hd = lane >> 3;                                   // 8 lanes per head (32 ch)
    float alpha = ew[(size_t)e * HEADS + hd] / esum[d * HEADS + hd];
    int c0 = lane * 4;
    const float4 hv = *(const float4*)(hh + (size_t)s * HID + c0);
    float* ob = outacc + (size_t)d * HID + c0;
    atomicAdd(ob + 0, hv.x * alpha);
    atomicAdd(ob + 1, hv.y * alpha);
    atomicAdd(ob + 2, hv.z * alpha);
    atomicAdd(ob + 3, hv.w * alpha);
}

// ---------------- layer epilogue: h = relu(out + bias) ---------------------
__global__ void k_layer_epilogue(const float* __restrict__ outacc, const float* __restrict__ bias,
                                 float* __restrict__ h, int n_nodes) {
    int idx = blockIdx.x * blockDim.x + threadIdx.x;
    if (idx >= n_nodes * HID) return;
    h[idx] = fmaxf(outacc[idx] + bias[idx % HID], 0.0f);
}

// ---------------- pooling ---------------------------------------------------
__global__ void k_pool_init(float* psum, float* pmax, float* cnt, int Bn) {
    int idx = blockIdx.x * blockDim.x + threadIdx.x;
    if (idx < Bn * HID) { psum[idx] = 0.0f; pmax[idx] = -3.0e38f; }
    if (idx < Bn) cnt[idx] = 0.0f;
}

__global__ void k_pool_acc(const float* __restrict__ h, const int* __restrict__ batch,
                           float* __restrict__ psum, float* __restrict__ pmax,
                           float* __restrict__ cnt, int n_nodes) {
    int idx = blockIdx.x * blockDim.x + threadIdx.x;
    if (idx >= n_nodes * HID) return;
    int n = idx / HID, c = idx % HID;
    int b = batch[n];
    float v = h[idx];
    atomicAdd(&psum[b * HID + c], v);
    atomicMaxF(&pmax[b * HID + c], v);
    if (c == 0) atomicAdd(&cnt[b], 1.0f);
}

__global__ void k_pool_final(const float* __restrict__ psum, const float* __restrict__ pmax,
                             const float* __restrict__ cnt, float* __restrict__ zin, int Bn) {
    int idx = blockIdx.x * blockDim.x + threadIdx.x;
    if (idx >= Bn * HID) return;
    int b = idx / HID, c = idx % HID;
    float ct = fmaxf(cnt[b], 1.0f);
    zin[b * POOLD + c] = psum[idx] / ct;
    zin[b * POOLD + HID + c] = pmax[idx];
}

// ---------------- trackster encoder (tiny MLP, 1 thread per graph) ---------
__global__ void k_ts(const float* __restrict__ ts, const float* __restrict__ w1,
                     const float* __restrict__ b1, const float* __restrict__ g,
                     const float* __restrict__ bln, const float* __restrict__ w2,
                     const float* __restrict__ b2, float* __restrict__ zin, int Bn) {
    int b = blockIdx.x * blockDim.x + threadIdx.x;
    if (b >= Bn) return;
    float hid[TSH];
    float t0 = ts[b * 3 + 0], t1 = ts[b * 3 + 1], t2 = ts[b * 3 + 2];
    float m = 0.f;
    for (int j = 0; j < TSH; j++) {
        hid[j] = b1[j] + t0 * w1[j] + t1 * w1[TSH + j] + t2 * w1[2 * TSH + j];
        m += hid[j];
    }
    m *= (1.0f / TSH);
    float var = 0.f;
    for (int j = 0; j < TSH; j++) { float dd = hid[j] - m; var += dd * dd; }
    var *= (1.0f / TSH);
    float rinv = rsqrtf(var + 1e-5f);
    for (int j = 0; j < TSH; j++)
        hid[j] = fmaxf((hid[j] - m) * rinv * g[j] + bln[j], 0.0f);
    for (int j2 = 0; j2 < TSH; j2++) {
        float acc = b2[j2];
        for (int j = 0; j < TSH; j++) acc += hid[j] * w2[j * TSH + j2];
        zin[b * POOLD + 2 * HID + j2] = acc;
    }
}

// ---------------- classifier: LN(320) -> 128 relu -> 8 ----------------------
__global__ void __launch_bounds__(128)
k_classifier(const float* __restrict__ zin, const float* __restrict__ g,
             const float* __restrict__ bln, const float* __restrict__ w1,
             const float* __restrict__ b1, const float* __restrict__ w2,
             const float* __restrict__ b2, float* __restrict__ out) {
    __shared__ float sz[POOLD];
    __shared__ float sh[HID];
    __shared__ float red[128];
    const int b = blockIdx.x, t = threadIdx.x;
    const float* z = zin + b * POOLD;

    float ls = 0.f;
    for (int i = t; i < POOLD; i += 128) ls += z[i];
    red[t] = ls; __syncthreads();
    for (int s = 64; s > 0; s >>= 1) { if (t < s) red[t] += red[t + s]; __syncthreads(); }
    float mean = red[0] * (1.0f / POOLD);
    __syncthreads();

    float lv = 0.f;
    for (int i = t; i < POOLD; i += 128) { float dd = z[i] - mean; lv += dd * dd; }
    red[t] = lv; __syncthreads();
    for (int s = 64; s > 0; s >>= 1) { if (t < s) red[t] += red[t + s]; __syncthreads(); }
    float rinv = rsqrtf(red[0] * (1.0f / POOLD) + 1e-5f);
    __syncthreads();

    for (int i = t; i < POOLD; i += 128)
        sz[i] = (z[i] - mean) * rinv * g[i] + bln[i];
    __syncthreads();

    float acc = b1[t];
    for (int i = 0; i < POOLD; i++) acc += sz[i] * w1[i * HID + t];
    sh[t] = fmaxf(acc, 0.0f);
    __syncthreads();

    if (t < NCLS) {
        float o = b2[t];
        for (int j = 0; j < HID; j++) o += sh[j] * w2[j * NCLS + t];
        out[b * NCLS + t] = o;
    }
}

extern "C" void kernel_launch(void* const* d_in, const int* in_sizes, int n_in,
                              void* d_out, int out_size, void* d_ws, size_t ws_size,
                              hipStream_t stream) {
    const float* x      = (const float*)d_in[0];
    const int*   ei     = (const int*)  d_in[1];
    const int*   batch  = (const int*)  d_in[2];
    const float* ts     = (const float*)d_in[3];
    const float* enc_w  = (const float*)d_in[4];
    const float* enc_b  = (const float*)d_in[5];
    const float* gat_w  = (const float*)d_in[6];
    const float* gat_as = (const float*)d_in[7];
    const float* gat_ad = (const float*)d_in[8];
    const float* gat_b  = (const float*)d_in[9];
    const float* ts_w1  = (const float*)d_in[10];
    const float* ts_b1  = (const float*)d_in[11];
    const float* ts_g   = (const float*)d_in[12];
    const float* ts_bln = (const float*)d_in[13];
    const float* ts_w2  = (const float*)d_in[14];
    const float* ts_b2  = (const float*)d_in[15];
    const float* cls_g  = (const float*)d_in[16];
    const float* cls_bln= (const float*)d_in[17];
    const float* cls_w1 = (const float*)d_in[18];
    const float* cls_b1 = (const float*)d_in[19];
    const float* cls_w2 = (const float*)d_in[20];
    const float* cls_b2 = (const float*)d_in[21];
    float* out = (float*)d_out;

    const int Nn = in_sizes[0] / 4;    // nodes
    const int E  = in_sizes[1] / 2;    // directed edges (pre self-loop)
    const int Bn = in_sizes[3] / 3;    // graphs
    const int Etot = E + Nn;

    // carve workspace (f32)
    float* p = (float*)d_ws;
    float* h      = p; p += (size_t)Nn * HID;
    float* hh     = p; p += (size_t)Nn * HID;
    float* outacc = p; p += (size_t)Nn * HID;
    float* a_s    = p; p += (size_t)Nn * HEADS;
    float* a_d    = p; p += (size_t)Nn * HEADS;
    float* emax   = p; p += (size_t)Nn * HEADS;
    float* esum   = p; p += (size_t)Nn * HEADS;
    float* ew     = p; p += (size_t)Etot * HEADS;
    float* psum   = p; p += (size_t)Bn * HID;
    float* pmax   = p; p += (size_t)Bn * HID;
    float* cnt    = p; p += (size_t)Bn;
    float* zin    = p; p += (size_t)Bn * POOLD;

    k_encoder<<<(Nn * HID + 255) / 256, 256, 0, stream>>>(x, enc_w, enc_b, h, Nn);

    for (int l = 0; l < LAYERS; l++) {
        k_gemm_wmma<<<(Nn + RPB - 1) / RPB, 128, 0, stream>>>(
            h, gat_w + (size_t)l * HID * HID, hh, Nn);
        k_attn_coef<<<(Nn * HEADS + 255) / 256, 256, 0, stream>>>(
            hh, gat_as + l * HEADS * CH, gat_ad + l * HEADS * CH, a_s, a_d, Nn);
        k_init_layer<<<(Nn * HID + 255) / 256, 256, 0, stream>>>(outacc, emax, esum, Nn);
        k_edge_max<<<(Etot + 255) / 256, 256, 0, stream>>>(ei, E, Nn, a_s, a_d, emax);
        k_edge_expsum<<<(Etot + 255) / 256, 256, 0, stream>>>(ei, E, Nn, a_s, a_d, emax, ew, esum);
        k_edge_aggregate<<<(Etot + 7) / 8, 256, 0, stream>>>(ei, E, Nn, hh, ew, esum, outacc);
        k_layer_epilogue<<<(Nn * HID + 255) / 256, 256, 0, stream>>>(outacc, gat_b + l * HID, h, Nn);
    }

    k_pool_init<<<(Bn * HID + 255) / 256, 256, 0, stream>>>(psum, pmax, cnt, Bn);
    k_pool_acc<<<(Nn * HID + 255) / 256, 256, 0, stream>>>(h, batch, psum, pmax, cnt, Nn);
    k_pool_final<<<(Bn * HID + 255) / 256, 256, 0, stream>>>(psum, pmax, cnt, zin, Bn);
    k_ts<<<(Bn + 63) / 64, 64, 0, stream>>>(ts, ts_w1, ts_b1, ts_g, ts_bln, ts_w2, ts_b2, zin, Bn);
    k_classifier<<<Bn, 128, 0, stream>>>(zin, cls_g, cls_bln, cls_w1, cls_b1, cls_w2, cls_b2, out);
}
